// SwitchLinear_61933428408855
// MI455X (gfx1250) — matrix operand order repaired and checked by reading, not compiled, not run on hardware
//
#include <hip/hip_runtime.h>
#include <hip/hip_bf16.h>
#include <stdint.h>

// ---------------------------------------------------------------------------
// SwitchLinear (MoE linear) for MI455X / gfx1250.
//   out[n] = W[idx[n]] @ x[n] + b[idx[n]]   (N=2048 tokens, 512x512, 16 experts)
//
// Bucket tokens by expert, then grouped GEMM with v_wmma_f32_16x16x32_bf16
// using a split-bf16 (hi+lo) decomposition of both x and W:
//   x*W ~= xh*Wh + xh*Wl + xl*Wh      (xl*Wl term ~2^-16, dropped)
// M-tile = 32 tokens per block (2 wmma m-tiles) so each B fragment feeds
// 6 WMMAs -> W-side L2 traffic ~64 MB, approaching the 24 MB HBM floor.
// ---------------------------------------------------------------------------

#define NUM_EXPERTS 16
#define IN_F  512
#define OUT_F 512
#define N_TOK 2048
#define M_TILE 32                      // tokens per block (2 wmma m-tiles)

typedef __attribute__((ext_vector_type(16))) __bf16 v16bf;
typedef __attribute__((ext_vector_type(8)))  float  v8f;

// LDS row stride for the 32x512 bf16 A-tiles: 512 + 8 pad elements so the
// 16 lanes of a half-wave (stride 1040B = 260 words = 4 banks apart) fan
// out across all 64 LDS banks on the 16B fragment reads.
#define A_STRIDE 520

union Frag {
    uint4  q[2];
    v16bf  v;
};

// ---------------------------------------------------------------------------
__global__ void zero_counts_kernel(int* counts) {
    if (threadIdx.x < NUM_EXPERTS) counts[threadIdx.x] = 0;
}

// One block per token (128 threads): scatter token id into its expert bucket
// and convert the token's x row to bf16 hi/lo.
__global__ void prep_tokens_kernel(const float* __restrict__ x,
                                   const long long* __restrict__ idx,
                                   __bf16* __restrict__ xhi,
                                   __bf16* __restrict__ xlo,
                                   int* __restrict__ counts,
                                   int* __restrict__ bucket) {
    const int t = blockIdx.x;
    if (threadIdx.x == 0) {
        int e = (int)idx[t];
        int pos = atomicAdd(&counts[e], 1);
        bucket[e * N_TOK + pos] = t;
    }
    // 512 floats per row, 128 threads -> one float4 each.
    const int c4 = threadIdx.x;             // 0..127
    const float4 v = ((const float4*)(x + (size_t)t * IN_F))[c4];
    __bf16 h0 = (__bf16)v.x, h1 = (__bf16)v.y, h2 = (__bf16)v.z, h3 = (__bf16)v.w;
    __bf16 l0 = (__bf16)(v.x - (float)h0);
    __bf16 l1 = (__bf16)(v.y - (float)h1);
    __bf16 l2 = (__bf16)(v.z - (float)h2);
    __bf16 l3 = (__bf16)(v.w - (float)h3);
    union { __bf16 b[4]; uint2 u; } ph, pl;
    ph.b[0] = h0; ph.b[1] = h1; ph.b[2] = h2; ph.b[3] = h3;
    pl.b[0] = l0; pl.b[1] = l1; pl.b[2] = l2; pl.b[3] = l3;
    ((uint2*)(xhi + (size_t)t * IN_F))[c4] = ph.u;
    ((uint2*)(xlo + (size_t)t * IN_F))[c4] = pl.u;
}

// One thread per float4: fp32 -> bf16 hi/lo split of all expert weights.
__global__ void convert_w_kernel(const float* __restrict__ W,
                                 __bf16* __restrict__ whi,
                                 __bf16* __restrict__ wlo,
                                 int n4) {
    int i = blockIdx.x * blockDim.x + threadIdx.x;
    if (i >= n4) return;
    const float4 v = ((const float4*)W)[i];
    __bf16 h0 = (__bf16)v.x, h1 = (__bf16)v.y, h2 = (__bf16)v.z, h3 = (__bf16)v.w;
    __bf16 l0 = (__bf16)(v.x - (float)h0);
    __bf16 l1 = (__bf16)(v.y - (float)h1);
    __bf16 l2 = (__bf16)(v.z - (float)h2);
    __bf16 l3 = (__bf16)(v.w - (float)h3);
    union { __bf16 b[4]; uint2 u; } ph, pl;
    ph.b[0] = h0; ph.b[1] = h1; ph.b[2] = h2; ph.b[3] = h3;
    pl.b[0] = l0; pl.b[1] = l1; pl.b[2] = l2; pl.b[3] = l3;
    ((uint2*)whi)[i] = ph.u;
    ((uint2*)wlo)[i] = pl.u;
}

// ---------------------------------------------------------------------------
// Grouped GEMM: block = (32-token tile) x (expert). 256 threads = 8 wave32.
// Each wave computes a 32x64 output slab: 2 m-tiles x 4 n-tiles of 16x16.
// K=512 in 16 steps of v_wmma_f32_16x16x32_bf16 (x3 for the hi/lo split,
// x2 m-tiles sharing each B fragment).
// ---------------------------------------------------------------------------
__global__ __launch_bounds__(256)
void moe_gemm_kernel(const __bf16* __restrict__ Whi,
                     const __bf16* __restrict__ Wlo,
                     const __bf16* __restrict__ xhi,
                     const __bf16* __restrict__ xlo,
                     const float*  __restrict__ bias,
                     const int*    __restrict__ counts,
                     const int*    __restrict__ bucket,
                     float*        __restrict__ out) {
    const int e     = blockIdx.y;
    const int tile  = blockIdx.x;
    const int count = counts[e];
    const int base  = tile * M_TILE;
    if (base >= count) return;                 // uniform: EXEC stays all-ones

    __shared__ __align__(16) __bf16 sAhi[M_TILE * A_STRIDE];
    __shared__ __align__(16) __bf16 sAlo[M_TILE * A_STRIDE];
    __shared__ int sTok[M_TILE];

    const int tid = threadIdx.x;
    if (tid < M_TILE)
        sTok[tid] = (base + tid < count) ? bucket[e * N_TOK + base + tid] : 0;

    // Gather 32 token rows (bf16 hi/lo) into LDS: thread -> (row, 64-elem chunk).
    {
        const int r = tid >> 3;                // 0..31
        const int c = (tid & 7) * 64;          // 0..448
        const bool valid = (base + r) < count;
        const int tok = valid ? bucket[e * N_TOK + base + r] : 0;
        const uint4* srcH = (const uint4*)(xhi + (size_t)tok * IN_F + c);
        const uint4* srcL = (const uint4*)(xlo + (size_t)tok * IN_F + c);
        uint4* dstH = (uint4*)(sAhi + r * A_STRIDE + c);
        uint4* dstL = (uint4*)(sAlo + r * A_STRIDE + c);
        const uint4 z = make_uint4(0u, 0u, 0u, 0u);
#pragma unroll
        for (int j = 0; j < 8; ++j) {
            dstH[j] = valid ? srcH[j] : z;
            dstL[j] = valid ? srcL[j] : z;
        }
    }
    __syncthreads();

    const int lane = tid & 31;
    const int wv   = tid >> 5;                 // wave id 0..7 -> N slab
    const int nl   = lane & 15;                // row (A) / column (B,C)
    const int h    = lane >> 4;                // lane half

    const __bf16* BhiE = Whi + (size_t)e * OUT_F * IN_F;
    const __bf16* BloE = Wlo + (size_t)e * OUT_F * IN_F;

    v8f   acc[8];                              // [mt*4 + nt]
    float bv[4];
#pragma unroll
    for (int i = 0; i < 8; ++i) acc[i] = {};
#pragma unroll
    for (int nt = 0; nt < 4; ++nt)
        bv[nt] = bias[e * OUT_F + wv * 64 + nt * 16 + nl];

    for (int kb = 0; kb < IN_F / 32; ++kb) {
        // A fragments (16x32 bf16, ISA 7.12.2): lane (M=nl, half h) holds
        // K = [h*8, h*8+8) in v0..3 and K = [16+h*8, ...) in v4..7.
        Frag aH[2], aL[2];
#pragma unroll
        for (int mt = 0; mt < 2; ++mt) {
            const __bf16* paH = sAhi + (mt * 16 + nl) * A_STRIDE + kb * 32;
            const __bf16* paL = sAlo + (mt * 16 + nl) * A_STRIDE + kb * 32;
            aH[mt].q[0] = *(const uint4*)(paH + h * 8);
            aH[mt].q[1] = *(const uint4*)(paH + 16 + h * 8);
            aL[mt].q[0] = *(const uint4*)(paL + h * 8);
            aL[mt].q[1] = *(const uint4*)(paL + 16 + h * 8);
        }

#pragma unroll
        for (int nt = 0; nt < 4; ++nt) {
            // B fragment (32x16 bf16): lane (N=nl, half h) holds K =
            // [kb*32 + h*8, +8) in v0..3 and [kb*32 + 16 + h*8, +8) in v4..7
            // (W is [out,in] row-major -> already K-contiguous = B^T feed).
            const size_t roff =
                (size_t)(wv * 64 + nt * 16 + nl) * IN_F + kb * 32 + h * 8;
            Frag bH, bL;
            bH.q[0] = *(const uint4*)(BhiE + roff);
            bH.q[1] = *(const uint4*)(BhiE + roff + 16);
            bL.q[0] = *(const uint4*)(BloE + roff);
            bL.q[1] = *(const uint4*)(BloE + roff + 16);

#pragma unroll
            for (int mt = 0; mt < 2; ++mt) {
                v8f c = acc[mt * 4 + nt];
                c = __builtin_amdgcn_wmma_f32_16x16x32_bf16(
                    false, aH[mt].v, false, bH.v, (short)0, c, false, false);
                c = __builtin_amdgcn_wmma_f32_16x16x32_bf16(
                    false, aH[mt].v, false, bL.v, (short)0, c, false, false);
                c = __builtin_amdgcn_wmma_f32_16x16x32_bf16(
                    false, aL[mt].v, false, bH.v, (short)0, c, false, false);
                acc[mt * 4 + nt] = c;
            }
        }
    }

    // Epilogue: C/D layout -> lane (N=nl, half h), VGPR i holds M = h*8 + i.
#pragma unroll
    for (int mt = 0; mt < 2; ++mt) {
#pragma unroll
        for (int i = 0; i < 8; ++i) {
            const int m = mt * 16 + h * 8 + i;
            if (base + m < count) {
                const int tok = sTok[m];
#pragma unroll
                for (int nt = 0; nt < 4; ++nt)
                    out[(size_t)tok * OUT_F + wv * 64 + nt * 16 + nl] =
                        acc[mt * 4 + nt][i] + bv[nt];
            }
        }
    }
}

// ---------------------------------------------------------------------------
extern "C" void kernel_launch(void* const* d_in, const int* in_sizes, int n_in,
                              void* d_out, int out_size, void* d_ws, size_t ws_size,
                              hipStream_t stream) {
    (void)in_sizes; (void)n_in; (void)out_size; (void)ws_size;

    const float*     x   = (const float*)d_in[0];
    const long long* idx = (const long long*)d_in[1];
    const float*     W   = (const float*)d_in[2];
    const float*     b   = (const float*)d_in[3];
    float*           out = (float*)d_out;

    // Workspace layout (~21.2 MB total)
    const size_t WN = (size_t)NUM_EXPERTS * OUT_F * IN_F;   // 4,194,304
    const size_t XN = (size_t)N_TOK * IN_F;                 // 1,048,576
    __bf16* Whi = (__bf16*)d_ws;
    __bf16* Wlo = Whi + WN;
    __bf16* Xhi = Wlo + WN;
    __bf16* Xlo = Xhi + XN;
    int* counts = (int*)(Xlo + XN);
    int* bucket = counts + 64;                               // pad to 256B

    zero_counts_kernel<<<1, 32, 0, stream>>>(counts);

    const int n4 = (int)(WN / 4);
    convert_w_kernel<<<(n4 + 255) / 256, 256, 0, stream>>>(W, Whi, Wlo, n4);

    prep_tokens_kernel<<<N_TOK, 128, 0, stream>>>(x, idx, Xhi, Xlo, counts, bucket);

    moe_gemm_kernel<<<dim3(N_TOK / M_TILE, NUM_EXPERTS), 256, 0, stream>>>(
        Whi, Wlo, Xhi, Xlo, b, counts, bucket, out);
}
